// FullyFusedMLP_25202868092982
// MI455X (gfx1250) — matrix operand order repaired and checked by reading, not compiled, not run on hardware
//
#include <hip/hip_runtime.h>

typedef __attribute__((ext_vector_type(16))) _Float16 v16h;
typedef __attribute__((ext_vector_type(8)))  _Float16 v8h;
typedef __attribute__((ext_vector_type(4)))  _Float16 v4h;
typedef __attribute__((ext_vector_type(8)))  float    v8f;
typedef __attribute__((ext_vector_type(4)))  float    v4f;

namespace {
// Geometry (fixed by the reference): 64 -> 128 -> 128 -> 128 -> 128 -> 64
constexpr int PW_IN = 72;   // padded row stride (halfs) for w_in   [128 x 64]
constexpr int PW    = 136;  // padded row stride (halfs) for 128-col weights
constexpr int PACT  = 136;  // padded activation row stride (halfs)

// LDS layout (offsets in halfs)
constexpr int OFF_W0   = 0;                      // w_in   128 x 72
constexpr int OFF_W1   = OFF_W0 + 128 * PW_IN;   // w_hid0 128 x 136
constexpr int OFF_W2   = OFF_W1 + 128 * PW;      // w_hid1
constexpr int OFF_W3   = OFF_W2 + 128 * PW;      // w_hid2
constexpr int OFF_WO   = OFF_W3 + 128 * PW;      // w_out  64 x 136
constexpr int OFF_ACT  = OFF_WO + 64 * PW;       // 8 waves x 16 x 136
constexpr int LDS_HALFS = OFF_ACT + 8 * 16 * PACT;
constexpr int BIAS_OFF_B = LDS_HALFS * 2;        // byte offset of fp32 bias block
constexpr size_t LDS_BYTES = (size_t)BIAS_OFF_B + 576 * 4;  // 177,408 B
}

// ---- fragment loads (layouts per CDNA5 ISA 7.12.2, wave32) ------------------

// A: 16x32 f16 tile of the activation matrix, K-chunk kc.
// lane m=lane&15; low half-wave holds K = kc*32 + {0..7, 16..23},
// high half-wave holds K = kc*32 + {8..15, 24..31}.
__device__ __forceinline__ v16h load_a_frag(const _Float16* act, int lane, int kc) {
  const int m     = lane & 15;
  const int kbase = kc * 32 + ((lane & 16) ? 8 : 0);
  const _Float16* p = act + m * PACT + kbase;
  v8h lo = *(const v8h*)(p);
  v8h hi = *(const v8h*)(p + 16);
  return __builtin_shufflevector(lo, hi, 0,1,2,3,4,5,6,7,8,9,10,11,12,13,14,15);
}

// B: 32x16 f16 tile; B[k][n] = W[nt*16+n][k]  (y = h @ W^T).
// lane n=lane&15; low half-wave holds K = kc*32 + {0..7, 16..23},
// high half-wave holds K = kc*32 + {8..15, 24..31} (matches A).
__device__ __forceinline__ v16h load_b_frag(const _Float16* w, int ws, int lane,
                                            int nt, int kc) {
  const int n    = lane & 15;
  const int koff = kc * 32 + ((lane & 16) ? 8 : 0);
  const _Float16* p = w + (nt * 16 + n) * ws + koff;
  v8h lo = *(const v8h*)(p);
  v8h hi = *(const v8h*)(p + 16);
  return __builtin_shufflevector(lo, hi, 0,1,2,3,4,5,6,7,8,9,10,11,12,13,14,15);
}

// ---- cooperative fp32 -> f16 weight staging into LDS ------------------------
template <int ROWS, int COLS, int DSTRIDE>
__device__ __forceinline__ void stage(const float* __restrict__ src,
                                      _Float16* dst, int tid) {
  for (int i = tid; i < ROWS * COLS; i += 256) {
    int r = i / COLS;
    int c = i - r * COLS;
    dst[r * DSTRIDE + c] = (_Float16)src[i];
  }
}

// ---- one hidden layer: act <- relu(act @ W^T + b), in place -----------------
// All A-fragments are hoisted to registers before any writeback, so in-place
// update is safe within the wave (same-wave LDS ops are in order).
template <int KDIM, int NTILES>
__device__ __forceinline__ void layer(_Float16* act, const _Float16* w, int ws,
                                      const float* bias, int lane) {
  constexpr int KC = KDIM / 32;
  v16h afr[KC];
#pragma unroll
  for (int kc = 0; kc < KC; ++kc) afr[kc] = load_a_frag(act, lane, kc);
  const int n     = lane & 15;
  const int mbase = (lane & 16) ? 8 : 0;
#pragma unroll
  for (int nt = 0; nt < NTILES; ++nt) {
    v8f c = {};
#pragma unroll
    for (int kc = 0; kc < KC; ++kc) {
      v16h b = load_b_frag(w, ws, lane, nt, kc);
      c = __builtin_amdgcn_wmma_f32_16x16x32_f16(false, afr[kc], false, b,
                                                 (short)0, c, false, false);
    }
    const float bv = bias[nt * 16 + n];
#pragma unroll
    for (int r = 0; r < 8; ++r) {
      float v = c[r] + bv;
      v = v > 0.0f ? v : 0.0f;                       // ReLU
      act[(mbase + r) * PACT + nt * 16 + n] = (_Float16)v;
    }
  }
}

// ---- output layer: out = act @ Wout^T + b (fp32, no activation) -------------
template <int KDIM, int NTILES>
__device__ __forceinline__ void layer_out(const _Float16* act,
                                          float* __restrict__ outp,
                                          const _Float16* w, int ws,
                                          const float* bias, int lane) {
  constexpr int KC = KDIM / 32;
  v16h afr[KC];
#pragma unroll
  for (int kc = 0; kc < KC; ++kc) afr[kc] = load_a_frag(act, lane, kc);
  const int n     = lane & 15;
  const int mbase = (lane & 16) ? 8 : 0;
#pragma unroll
  for (int nt = 0; nt < NTILES; ++nt) {
    v8f c = {};
#pragma unroll
    for (int kc = 0; kc < KC; ++kc) {
      v16h b = load_b_frag(w, ws, lane, nt, kc);
      c = __builtin_amdgcn_wmma_f32_16x16x32_f16(false, afr[kc], false, b,
                                                 (short)0, c, false, false);
    }
    const float bv = bias[nt * 16 + n];
#pragma unroll
    for (int r = 0; r < 8; ++r)
      outp[(mbase + r) * 64 + nt * 16 + n] = c[r] + bv;
  }
}

__global__ __launch_bounds__(256)
void FullyFusedMLP_25202868092982_kernel(
    const float* __restrict__ x,     const float* __restrict__ w_in,
    const float* __restrict__ b_in,  const float* __restrict__ w_hid,
    const float* __restrict__ b_hid, const float* __restrict__ w_out,
    const float* __restrict__ b_out, float* __restrict__ out, int nrows) {
  extern __shared__ __align__(16) char smem[];
  _Float16* sh = (_Float16*)smem;
  float*    sb = (float*)(smem + BIAS_OFF_B);
  const int tid = threadIdx.x;

  // Stage all weights (fp32 -> f16) + biases (fp32) into LDS once per block.
  stage<128, 64, PW_IN>(w_in, sh + OFF_W0, tid);
  stage<128, 128, PW>(w_hid,           sh + OFF_W1, tid);
  stage<128, 128, PW>(w_hid + 16384,   sh + OFF_W2, tid);
  stage<128, 128, PW>(w_hid + 32768,   sh + OFF_W3, tid);
  stage<64, 128, PW>(w_out,            sh + OFF_WO, tid);
  for (int i = tid; i < 128; i += 256) sb[i]       = b_in[i];
  for (int i = tid; i < 384; i += 256) sb[128 + i] = b_hid[i];
  for (int i = tid; i < 64;  i += 256) sb[512 + i] = b_out[i];
  __syncthreads();

  const int lane = tid & 31;
  const int wave = tid >> 5;
  _Float16* act = sh + OFF_ACT + wave * (16 * PACT);
  const int ntile_rows = nrows >> 4;         // 16-row tiles
  const int stride = gridDim.x * 8;          // 8 waves per block

  for (int tb = blockIdx.x * 8 + wave; tb < ntile_rows; tb += stride) {
    const float* xt = x + (size_t)tb * (16 * 64);
    __builtin_prefetch(xt + (size_t)stride * 16 * 64, 0, 3);

    // Load 16x64 fp32 input tile, convert to f16 into this wave's LDS tile.
#pragma unroll
    for (int e = 0; e < 8; ++e) {
      const int idx = lane + e * 32;          // 256 float4s total
      const int row = idx >> 4;
      const int col = (idx & 15) << 2;
      v4f f = *(const v4f*)(xt + row * 64 + col);
      v4h h;
      h[0] = (_Float16)f[0]; h[1] = (_Float16)f[1];
      h[2] = (_Float16)f[2]; h[3] = (_Float16)f[3];
      *(v4h*)(act + row * PACT + col) = h;
    }

    layer<64, 8>(act, sh + OFF_W0, PW_IN, sb + 0, lane);     // in-layer + ReLU
    layer<128, 8>(act, sh + OFF_W1, PW, sb + 128, lane);     // hidden 0
    layer<128, 8>(act, sh + OFF_W2, PW, sb + 256, lane);     // hidden 1
    layer<128, 8>(act, sh + OFF_W3, PW, sb + 384, lane);     // hidden 2
    layer_out<128, 4>(act, out + (size_t)tb * (16 * 64),
                      sh + OFF_WO, PW, sb + 512, lane);      // out-layer
  }
}

extern "C" void kernel_launch(void* const* d_in, const int* in_sizes, int n_in,
                              void* d_out, int out_size, void* d_ws, size_t ws_size,
                              hipStream_t stream) {
  const float* x     = (const float*)d_in[0];
  const float* w_in  = (const float*)d_in[1];
  const float* b_in  = (const float*)d_in[2];
  const float* w_hid = (const float*)d_in[3];
  const float* b_hid = (const float*)d_in[4];
  const float* w_out = (const float*)d_in[5];
  const float* b_out = (const float*)d_in[6];
  float* out = (float*)d_out;

  const int nrows  = in_sizes[0] / 64;
  const int ntiles = nrows >> 4;
  int blocks = (ntiles + 7) / 8;
  if (blocks > 1024) blocks = 1024;
  if (blocks < 1) blocks = 1;

  FullyFusedMLP_25202868092982_kernel<<<dim3(blocks), dim3(256), LDS_BYTES,
                                        stream>>>(
      x, w_in, b_in, w_hid, b_hid, w_out, b_out, out, nrows);
}
